// SparseAttention_26250840113200
// MI455X (gfx1250) — matrix operand order repaired and checked by reading, not compiled, not run on hardware
//
#include <hip/hip_runtime.h>
#include <hip/hip_bf16.h>
#include <stdint.h>

typedef __attribute__((ext_vector_type(16))) _Float16 v16h;
typedef __attribute__((ext_vector_type(8)))  _Float16 v8h;
typedef __attribute__((ext_vector_type(8)))  float    v8f;

// Problem constants (fixed by the reference)
// B=2, S=2048, D=1024, H=16, hd=64, WINDOW/2=64, n_keep=675

// ---------------- optional gfx1250 async global->LDS path ----------------
#if __has_builtin(__builtin_amdgcn_global_load_async_to_lds_b128)
#define USE_ASYNC_LDS 1
#else
#define USE_ASYNC_LDS 0
#endif

#if __has_builtin(__builtin_amdgcn_s_wait_asynccnt)
#define ASYNC_WAIT0() __builtin_amdgcn_s_wait_asynccnt(0)
#else
#define ASYNC_WAIT0() asm volatile("s_wait_asynccnt 0x0" ::: "memory")
#endif

#if USE_ASYNC_LDS
typedef int v4i_ __attribute__((vector_size(16)));
__device__ __forceinline__ void async_ld_b128(const void* gptr, void* lptr) {
  typedef __attribute__((address_space(1))) v4i_ GT;   // global (prints as __device__)
  typedef __attribute__((address_space(3))) v4i_ LT;   // LDS    (prints as __shared__)
  // global generic pointers are numerically identical to AS1 addresses;
  // LDS generic pointers carry the LDS offset in addr[31:0] (ISA aperture rule).
  GT* g = (GT*)(uintptr_t)gptr;
  LT* l = (LT*)(uint32_t)(uintptr_t)lptr;
  __builtin_amdgcn_global_load_async_to_lds_b128(g, l, 0, 0);
}
#endif

// ---------------- WMMA operand loaders (gfx1250 wave32 layouts) ----------------
// A-matrix 16x32 f16, row-major source, row stride ld (halfs).
// lanes 0-15: row=lane,    a[0..15]=K0..15
// lanes 16-31: row=lane-16, a[0..15]=K16..31
__device__ __forceinline__ v16h load_a16(const _Float16* __restrict__ base, int ld, int lane) {
  const int r  = lane & 15;
  const int hi = (lane >> 4) & 1;
  const _Float16* p = base + (size_t)r * ld + hi * 16;
  v8h lo = *(const v8h*)(p);
  v8h hh = *(const v8h*)(p + 8);
  v16h a;
#pragma unroll
  for (int i = 0; i < 8; ++i) { a[i] = lo[i]; a[i + 8] = hh[i]; }
  return a;
}

// B-matrix 32x16 f16 loaded from a row-major "B-transposed" tile (16 rows = out cols, K contiguous).
__device__ __forceinline__ v16h load_bt16(const _Float16* __restrict__ base, int ld, int lane) {
  const int c  = lane & 15;
  const int hi = (lane >> 4) & 1;
  return *(const v16h*)(base + (size_t)c * ld + hi * 16);
}

__device__ __forceinline__ v8f wmma_f16(v16h a, v16h b, v8f c) {
  return __builtin_amdgcn_wmma_f32_16x16x32_f16(false, a, false, b, (short)0, c, false, false);
}

// ---------------- K1: cast inputs to f16 ----------------
__global__ void k_cast(const float* __restrict__ x, const float* __restrict__ wqkv,
                       const float* __restrict__ wout,
                       _Float16* __restrict__ xh, _Float16* __restrict__ wqkvh,
                       _Float16* __restrict__ wouth) {
  int i = blockIdx.x * blockDim.x + threadIdx.x;
  const int n1 = 4096 * 1024, n2 = 3072 * 1024, n3 = 1024 * 1024;
  if (i < n1) xh[i] = (_Float16)x[i];
  else if (i < n1 + n2) wqkvh[i - n1] = (_Float16)wqkv[i - n1];
  else if (i < n1 + n2 + n3) wouth[i - n1 - n2] = (_Float16)wout[i - n1 - n2];
}

// ---------------- K2: per-(b,h) centroid ----------------
__global__ void k_centroid(const float* __restrict__ x, float* __restrict__ cent) {
  int bh = blockIdx.x;                 // 32
  int b = bh >> 4, h = bh & 15;
  int d = threadIdx.x;                 // 64
  float s = 0.f;
  const float* p = x + (size_t)b * 2048 * 1024 + h * 64 + d;
  for (int ss = 0; ss < 2048; ++ss) s += p[(size_t)ss * 1024];
  cent[bh * 64 + d] = s * (1.0f / 2048.0f);
}

// ---------------- K3: d2c (f32) and ||hs||^2 (f16 values, f32 accumulate) ----------------
__global__ void k_d2c_sq(const float* __restrict__ x, const _Float16* __restrict__ xh,
                         const float* __restrict__ cent,
                         float* __restrict__ d2c, float* __restrict__ sq) {
  int bh = blockIdx.x; int b = bh >> 4, h = bh & 15;
  __shared__ float sc[64];
  if (threadIdx.x < 64) sc[threadIdx.x] = cent[bh * 64 + threadIdx.x];
  __syncthreads();
  for (int s = threadIdx.x; s < 2048; s += 256) {
    const float* px = x + ((size_t)(b * 2048 + s)) * 1024 + h * 64;
    const _Float16* ph = xh + ((size_t)(b * 2048 + s)) * 1024 + h * 64;
    float a = 0.f, q = 0.f;
    for (int d = 0; d < 64; ++d) {
      float t = px[d] - sc[d]; a += t * t;
      float v = (float)ph[d];  q += v * v;
    }
    d2c[bh * 2048 + s] = sqrtf(a);
    sq[bh * 2048 + s]  = q;
  }
}

// ---------------- K4: QKV projection (WMMA GEMM 4096x3072x1024) ----------------
__global__ void k_qkv(const _Float16* __restrict__ xh, const _Float16* __restrict__ wh,
                      const float* __restrict__ bias,
                      _Float16* __restrict__ q, _Float16* __restrict__ k,
                      _Float16* __restrict__ vT) {
  const int lane = threadIdx.x & 31;
  const int wv = threadIdx.x >> 5;
  int task = blockIdx.x * 8 + wv;      // 12288 = 256 (tm) * 48 (tn)
  int tm = task % 256;
  int tn = task / 256;
  const _Float16* A = xh + (size_t)tm * 16 * 1024;
  v8f acc[4] = {};
  for (int kk = 0; kk < 1024; kk += 32) {
    __builtin_prefetch(A + kk + 128, 0, 3);   // WGP-scope prefetch of the A stream
    v16h a = load_a16(A + kk, 1024, lane);
#pragma unroll
    for (int t = 0; t < 4; ++t) {
      v16h bt = load_bt16(wh + (size_t)(tn * 64 + t * 16) * 1024 + kk, 1024, lane);
      acc[t] = wmma_f16(a, bt, acc[t]);
    }
  }
  const int hi = (lane >> 4) & 1;
#pragma unroll
  for (int t = 0; t < 4; ++t) {
    int n = tn * 64 + t * 16 + (lane & 15);
    float bn = bias[n];
    int which = n >> 10;
    int hh = (n >> 6) & 15;
    int d = n & 63;
#pragma unroll
    for (int r = 0; r < 8; ++r) {
      int m = tm * 16 + r + hi * 8;
      int b = m >> 11, s = m & 2047;
      _Float16 hv = (_Float16)(acc[t][r] + bn);
      size_t qoff = (((size_t)(b * 16 + hh) * 2048 + s) * 64) + d;
      if (which == 0)      q[qoff] = hv;
      else if (which == 1) k[qoff] = hv;
      else                 vT[(((size_t)(b * 16 + hh) * 64 + d) * 2048) + s] = hv;
    }
  }
}

// ---------------- K5: gram matrix (WMMA) + per-row 10th-smallest distance ----------------
__global__ void k_gram(const _Float16* __restrict__ xh, const float* __restrict__ sq,
                       float* __restrict__ mmd) {
  const int lane = threadIdx.x & 31;
  const int wv = threadIdx.x >> 5;
  int bh = blockIdx.x >> 3;   // 32
  int rb = blockIdx.x & 7;    // 8 row blocks of 256 rows
  int b = bh >> 4, h = bh & 15;
  __shared__ float s_sq[2048];
  __shared__ float s_st[8][32 * 17];
  for (int i = threadIdx.x; i < 2048; i += 256) s_sq[i] = sq[bh * 2048 + i];
  __syncthreads();
  int strip = rb * 256 + wv * 32;
  const _Float16* A0 = xh + ((size_t)(b * 2048 + strip)) * 1024 + h * 64;
  const _Float16* A1 = A0 + 16 * 1024;
  int row = strip + lane;
  float sqi = s_sq[row];
  float top[10];
#pragma unroll
  for (int i = 0; i < 10; ++i) top[i] = 3.0e38f;
  const int hi = (lane >> 4) & 1;
  for (int jt = 0; jt < 128; ++jt) {
    v8f c0 = {}, c1 = {};
    const _Float16* Bt = xh + ((size_t)(b * 2048 + jt * 16)) * 1024 + h * 64;
#pragma unroll
    for (int kb = 0; kb < 2; ++kb) {
      int kk = kb * 32;
      v16h a0 = load_a16(A0 + kk, 1024, lane);
      v16h a1 = load_a16(A1 + kk, 1024, lane);
      v16h bt = load_bt16(Bt + kk, 1024, lane);
      c0 = wmma_f16(a0, bt, c0);
      c1 = wmma_f16(a1, bt, c1);
    }
#pragma unroll
    for (int r = 0; r < 8; ++r) {
      s_st[wv][(r + hi * 8) * 17 + (lane & 15)]      = c0[r];
      s_st[wv][(r + 16 + hi * 8) * 17 + (lane & 15)] = c1[r];
    }
    __syncthreads();   // lane<->row transpose inside the wave's private LDS region
    for (int n = 0; n < 16; ++n) {
      float dot = s_st[wv][lane * 17 + n];
      float d2 = sqi + s_sq[jt * 16 + n] - 2.0f * dot;
      float dd = sqrtf(fmaxf(d2, 0.0f));
      if (dd < top[9]) {
#pragma unroll
        for (int t = 9; t >= 0; --t)
          if (top[t] > dd) top[t] = (t > 0 && top[t - 1] > dd) ? top[t - 1] : dd;
      }
    }
  }
  mmd[bh * 2048 + row] = top[9];
}

// ---------------- K6: tls = 0.7*lifetime + 0.3*mmd_norm ----------------
__global__ void k_tls(const float* __restrict__ d2c, const float* __restrict__ mmd,
                      float* __restrict__ tls) {
  int bh = blockIdx.x;
  __shared__ float red[256];
  float mx = 0.f;
  for (int s = threadIdx.x; s < 2048; s += 256) mx = fmaxf(mx, d2c[bh * 2048 + s]);
  red[threadIdx.x] = mx; __syncthreads();
  for (int o = 128; o > 0; o >>= 1) {
    if (threadIdx.x < o) red[threadIdx.x] = fmaxf(red[threadIdx.x], red[threadIdx.x + o]);
    __syncthreads();
  }
  float maxd = red[0];
  __syncthreads();
  mx = 0.f;
  for (int s = threadIdx.x; s < 2048; s += 256) mx = fmaxf(mx, mmd[bh * 2048 + s]);
  red[threadIdx.x] = mx; __syncthreads();
  for (int o = 128; o > 0; o >>= 1) {
    if (threadIdx.x < o) red[threadIdx.x] = fmaxf(red[threadIdx.x], red[threadIdx.x + o]);
    __syncthreads();
  }
  float maxm = red[0];
  __syncthreads();
  float id = 1.0f / (maxd + 1e-8f), im = 1.0f / (maxm + 1e-8f);
  for (int s = threadIdx.x; s < 2048; s += 256)
    tls[bh * 2048 + s] = 0.7f * d2c[bh * 2048 + s] * id + 0.3f * mmd[bh * 2048 + s] * im;
}

// ---------------- K7: top-675 per (b,h) via bitonic sort, jax-style tie-break ----------------
__global__ void k_topk(const float* __restrict__ tls, int* __restrict__ lm) {
  int bh = blockIdx.x;
  __shared__ float v[2048];
  __shared__ float s_thr;
  __shared__ int s_cnt;
  for (int i = threadIdx.x; i < 2048; i += 1024) v[i] = tls[bh * 2048 + i];
  __syncthreads();
  for (int k = 2; k <= 2048; k <<= 1) {
    for (int j = k >> 1; j > 0; j >>= 1) {
      for (int e = threadIdx.x; e < 2048; e += 1024) {
        int ix = e ^ j;
        if (ix > e) {
          float a = v[e], bb = v[ix];
          bool up = ((e & k) == 0);
          if ((a > bb) == up) { v[e] = bb; v[ix] = a; }
        }
      }
      __syncthreads();
    }
  }
  if (threadIdx.x == 0) {      // ascending sort; 675th largest = v[2048-675]
    float thr = v[2048 - 675];
    int c = 0;
    for (int i = 2047; i >= 0 && v[i] > thr; --i) c++;
    s_thr = thr; s_cnt = c;
  }
  __syncthreads();
  float thr = s_thr;
  for (int i = threadIdx.x; i < 2048; i += 1024)
    lm[bh * 2048 + i] = (tls[bh * 2048 + i] > thr) ? 1 : 0;
  __syncthreads();
  if (threadIdx.x == 0) {      // fill ties in ascending index order (jax top_k tie-break)
    int rem = 675 - s_cnt;
    for (int i = 0; i < 2048 && rem > 0; ++i)
      if (tls[bh * 2048 + i] == thr) { lm[bh * 2048 + i] = 1; rem--; }
  }
}

// ---------------- K8: scores = QK^T (WMMA) + mask + raw write + online (m,l) ----------------
__global__ void k_scores(const _Float16* __restrict__ q, const _Float16* __restrict__ k,
                         const int* __restrict__ lm, float* __restrict__ attn,
                         float* __restrict__ mrow, float* __restrict__ lrow) {
  const int lane = threadIdx.x & 31;
  const int wv = threadIdx.x >> 5;
  int bh = blockIdx.x >> 3;
  int rb = blockIdx.x & 7;
  __shared__ int s_lm[2048];
  __shared__ float s_st[8][32 * 17];
  for (int i = threadIdx.x; i < 2048; i += 256) s_lm[i] = lm[bh * 2048 + i];
  __syncthreads();
  int strip = rb * 256 + wv * 32;
  const _Float16* Qb = q + ((size_t)bh * 2048 + strip) * 64;
  const _Float16* Kb = k + (size_t)bh * 2048 * 64;
  int row = strip + lane;
  float m = -3.0e38f, l = 0.f;
  const int hi = (lane >> 4) & 1;
  const float scale = 0.125f;   // 1/sqrt(64)
  for (int jt = 0; jt < 128; ++jt) {
    v8f c0 = {}, c1 = {};
#pragma unroll
    for (int kb = 0; kb < 2; ++kb) {
      int kk = kb * 32;
      v16h a0 = load_a16(Qb + kk, 64, lane);
      v16h a1 = load_a16(Qb + 16 * 64 + kk, 64, lane);
      v16h bt = load_bt16(Kb + (size_t)jt * 16 * 64 + kk, 64, lane);
      c0 = wmma_f16(a0, bt, c0);
      c1 = wmma_f16(a1, bt, c1);
    }
    int col = jt * 16 + (lane & 15);
    bool lmc = (s_lm[col] != 0);
#pragma unroll
    for (int r = 0; r < 8; ++r) {
      {
        int rr = strip + r + hi * 8;
        int rel = col - rr;
        bool ok = lmc || (rel >= -64 && rel <= 64);
        float sc = ok ? c0[r] * scale : -10000.0f;
        attn[((size_t)bh * 2048 + rr) * 2048 + col] = sc;
        s_st[wv][(r + hi * 8) * 17 + (lane & 15)] = sc;
      }
      {
        int rr = strip + 16 + r + hi * 8;
        int rel = col - rr;
        bool ok = lmc || (rel >= -64 && rel <= 64);
        float sc = ok ? c1[r] * scale : -10000.0f;
        attn[((size_t)bh * 2048 + rr) * 2048 + col] = sc;
        s_st[wv][(r + 16 + hi * 8) * 17 + (lane & 15)] = sc;
      }
    }
    __syncthreads();
    float vals[16];
    float tm = -3.0e38f;
#pragma unroll
    for (int n = 0; n < 16; ++n) { vals[n] = s_st[wv][lane * 17 + n]; tm = fmaxf(tm, vals[n]); }
    if (tm > m) { l *= expf(m - tm); m = tm; }
    float ls = 0.f;
#pragma unroll
    for (int n = 0; n < 16; ++n) ls += expf(vals[n] - m);
    l += ls;
  }
  mrow[(size_t)bh * 2048 + row] = m;
  lrow[(size_t)bh * 2048 + row] = l;
}

// ---------------- K9: fused normalize+write attn, then attn @ V (WMMA) -> ctx f16 ----------
// All 8 waves of a block share one (b,h): the 64x32 V-tile per k-step is staged
// global->LDS once per block via async-to-LDS (double-buffered), instead of 8x per wave.
__global__ void k_av(const float* __restrict__ mrow, const float* __restrict__ lrow,
                     const _Float16* __restrict__ vT, float* __restrict__ attn,
                     _Float16* __restrict__ ctx) {
  const int lane = threadIdx.x & 31;
  const int wv = threadIdx.x >> 5;
  const int tid = threadIdx.x;
  int task = blockIdx.x * 8 + wv;   // 4096 = 32 bh * 128 strips; 8 tasks/block share bh
  int bh = task >> 7;
  int st = task & 127;
  int b = bh >> 4, h = bh & 15;
  int strip = st * 16;
  int row = strip + (lane & 15);
  const int hi = (lane >> 4) & 1;
  float mi = mrow[(size_t)bh * 2048 + row];
  float inv = 1.0f / lrow[(size_t)bh * 2048 + row];
  v8f acc[4] = {};
  float* ap_base = attn + ((size_t)bh * 2048 + row) * 2048;
  const _Float16* Vb = vT + (size_t)bh * 64 * 2048;
#if USE_ASYNC_LDS
  __shared__ alignas(32) _Float16 s_v[2][64 * 32];   // 8 KB double buffer
  // thread tid moves 16B: V row = tid>>2, 8-half chunk = tid&3 -> LDS row-major [64][32]
  {
    const _Float16* g0 = Vb + (size_t)(tid >> 2) * 2048 + (tid & 3) * 8;
    async_ld_b128(g0, &s_v[0][tid * 8]);
  }
  for (int it = 0; it < 64; ++it) {
    const int kk = it * 32;
    const int buf = it & 1;
    ASYNC_WAIT0();        // this wave's async copies done
    __syncthreads();      // whole tile (all waves' chunks) visible
    if (it + 1 < 64) {    // overlap next tile's copy with this step's compute
      const _Float16* g = Vb + (size_t)(tid >> 2) * 2048 + (it + 1) * 32 + (tid & 3) * 8;
      async_ld_b128(g, &s_v[(it + 1) & 1][tid * 8]);
    }
    float* p0 = ap_base + kk + hi * 16;
    v16h a;
#pragma unroll
    for (int i = 0; i < 8; ++i) {
      float pv = expf(p0[i] - mi) * inv;
      p0[i] = pv;
      a[i] = (_Float16)pv;
    }
    float* p1 = p0 + 8;
#pragma unroll
    for (int i = 0; i < 8; ++i) {
      float pv = expf(p1[i] - mi) * inv;
      p1[i] = pv;
      a[i + 8] = (_Float16)pv;
    }
    const _Float16* vbuf = &s_v[buf][0];
#pragma unroll
    for (int t = 0; t < 4; ++t) {
      v16h bt = *(const v16h*)(vbuf + (t * 16 + (lane & 15)) * 32 + hi * 16);
      acc[t] = wmma_f16(a, bt, acc[t]);
    }
    __syncthreads();      // readers done before this buffer is refilled at it+2
  }
#else
  for (int kk = 0; kk < 2048; kk += 32) {
    float* p0 = ap_base + kk + hi * 16;
    v16h a;
#pragma unroll
    for (int i = 0; i < 8; ++i) {
      float pv = expf(p0[i] - mi) * inv;
      p0[i] = pv;
      a[i] = (_Float16)pv;
    }
    float* p1 = p0 + 8;
#pragma unroll
    for (int i = 0; i < 8; ++i) {
      float pv = expf(p1[i] - mi) * inv;
      p1[i] = pv;
      a[i + 8] = (_Float16)pv;
    }
#pragma unroll
    for (int t = 0; t < 4; ++t) {
      v16h bt = load_bt16(Vb + (size_t)(t * 16) * 2048 + kk, 2048, lane);
      acc[t] = wmma_f16(a, bt, acc[t]);
    }
  }
#endif
#pragma unroll
  for (int t = 0; t < 4; ++t) {
    int d = t * 16 + (lane & 15);
#pragma unroll
    for (int r = 0; r < 8; ++r) {
      int s = strip + r + hi * 8;
      ctx[((size_t)(b * 2048 + s)) * 1024 + h * 64 + d] = (_Float16)acc[t][r];
    }
  }
}

// ---------------- K10: output projection (WMMA GEMM 4096x1024x1024) ----------------
__global__ void k_out(const _Float16* __restrict__ ctx, const _Float16* __restrict__ wo,
                      const float* __restrict__ bias, float* __restrict__ out) {
  const int lane = threadIdx.x & 31;
  const int wv = threadIdx.x >> 5;
  int task = blockIdx.x * 8 + wv;  // 4096 = 256 tm * 16 tn
  int tm = task & 255;
  int tn = task >> 8;
  const _Float16* A = ctx + (size_t)tm * 16 * 1024;
  v8f acc[4] = {};
  for (int kk = 0; kk < 1024; kk += 32) {
    __builtin_prefetch(A + kk + 128, 0, 3);
    v16h a = load_a16(A + kk, 1024, lane);
#pragma unroll
    for (int t = 0; t < 4; ++t) {
      v16h bt = load_bt16(wo + (size_t)(tn * 64 + t * 16) * 1024 + kk, 1024, lane);
      acc[t] = wmma_f16(a, bt, acc[t]);
    }
  }
  const int hi = (lane >> 4) & 1;
#pragma unroll
  for (int t = 0; t < 4; ++t) {
    int n = tn * 64 + t * 16 + (lane & 15);
    float bn = bias[n];
#pragma unroll
    for (int r = 0; r < 8; ++r) {
      int m = tm * 16 + r + hi * 8;
      out[(size_t)m * 1024 + n] = acc[t][r] + bn;
    }
  }
}

// ---------------- Launch ----------------
extern "C" void kernel_launch(void* const* d_in, const int* in_sizes, int n_in,
                              void* d_out, int out_size, void* d_ws, size_t ws_size,
                              hipStream_t stream) {
  const float* x      = (const float*)d_in[0];   // (2,2048,1024)
  const float* qkv_w  = (const float*)d_in[1];   // (3072,1024)
  const float* qkv_b  = (const float*)d_in[2];   // (3072)
  const float* out_w  = (const float*)d_in[3];   // (1024,1024)
  const float* out_b  = (const float*)d_in[4];   // (1024)

  float* out0 = (float*)d_out;                   // (2,2048,1024)
  float* attn = out0 + (size_t)4096 * 1024;      // (2,16,2048,2048)

  // workspace layout (~52.2 MB total)
  char* w = (char*)d_ws;
  _Float16* xh    = (_Float16*)(w + 0);                 //  8,388,608
  _Float16* wqkvh = (_Float16*)(w + 8388608);           //  6,291,456
  _Float16* wouth = (_Float16*)(w + 14680064);          //  2,097,152
  _Float16* qh    = (_Float16*)(w + 16777216);          //  8,388,608
  _Float16* kh    = (_Float16*)(w + 25165824);          //  8,388,608
  _Float16* vT    = (_Float16*)(w + 33554432);          //  8,388,608
  _Float16* ctx   = (_Float16*)(w + 41943040);          //  8,388,608
  float* sq    = (float*)(w + 50331648);                //    262,144
  float* d2c   = (float*)(w + 50593792);
  float* mmd   = (float*)(w + 50855936);
  float* tls   = (float*)(w + 51118080);
  float* mrowp = (float*)(w + 51380224);
  float* lrowp = (float*)(w + 51642368);
  float* cent  = (float*)(w + 51904512);                //      8,192
  int*   lm    = (int*)  (w + 51920896);                //    262,144

  k_cast    <<<49152, 256, 0, stream>>>(x, qkv_w, out_w, xh, wqkvh, wouth);
  k_centroid<<<32,    64,  0, stream>>>(x, cent);
  k_d2c_sq  <<<32,    256, 0, stream>>>(x, xh, cent, d2c, sq);
  k_qkv     <<<1536,  256, 0, stream>>>(xh, wqkvh, qkv_b, qh, kh, vT);
  k_gram    <<<256,   256, 0, stream>>>(xh, sq, mmd);
  k_tls     <<<32,    256, 0, stream>>>(d2c, mmd, tls);
  k_topk    <<<32,   1024, 0, stream>>>(tls, lm);
  k_scores  <<<256,   256, 0, stream>>>(qh, kh, lm, attn, mrowp, lrowp);
  k_av      <<<512,   256, 0, stream>>>(mrowp, lrowp, vT, attn, ctx);
  k_out     <<<512,   256, 0, stream>>>(ctx, wouth, out_b, out0);
}
